// Attention_8847632630173
// MI455X (gfx1250) — matrix operand order repaired and checked by reading, not compiled, not run on hardware
//
#include <hip/hip_runtime.h>
#include <hip/hip_bf16.h>

typedef __bf16          v16bf  __attribute__((ext_vector_type(16)));
typedef __bf16          v8bf   __attribute__((ext_vector_type(8)));
typedef float           v8f    __attribute__((ext_vector_type(8)));
typedef float           f32x8  __attribute__((ext_vector_type(8)));
typedef float           f32x16 __attribute__((ext_vector_type(16)));
typedef unsigned short  u16x8  __attribute__((ext_vector_type(8)));
typedef unsigned short  u16x16 __attribute__((ext_vector_type(16)));

#define T_SEQ   2048
#define HEAD_D  64
#define BLOCK_M 128
#define BLOCK_N 128
#define NWAVES  8
#define NEG_BIG (-3.0e38f)

#define WMMA_BF16(A, B, C) \
  __builtin_amdgcn_wmma_f32_16x16x32_bf16(false, (A), false, (B), (short)0, (C), false, false)

// One DPP16 max-combine step: x = max(x, lane-permuted x). CTRL must be literal.
#define DPP_MAXF(x, ctrl)                                                      \
  do {                                                                         \
    int _xi = __builtin_bit_cast(int, (x));                                    \
    int _xr = __builtin_amdgcn_update_dpp(_xi, _xi, (ctrl), 0xf, 0xf, true);   \
    (x) = fmaxf((x), __builtin_bit_cast(float, _xr));                          \
  } while (0)

// f32 -> bf16 bits using the native conversion (RNE).
__device__ __forceinline__ unsigned short f2bf(float x) {
  union { __bf16 h; unsigned short u; } c;
  c.h = (__bf16)x;
  return c.u;
}

// Gather two 8-half chunks (16B each) into one A-operand v16bf.
__device__ __forceinline__ v16bf lds_a_gather(const unsigned short* p, int lo, int hi) {
  u16x8 a = *(const u16x8*)(p + lo);
  u16x8 b = *(const u16x8*)(p + hi);
  u16x16 t = __builtin_shufflevector(a, b, 0,1,2,3,4,5,6,7,8,9,10,11,12,13,14,15);
  return __builtin_bit_cast(v16bf, t);
}

// Contiguous 16 halfs (32B) -> B-operand v16bf (lane = one K row, elems = N).
__device__ __forceinline__ v16bf lds_b_row(const unsigned short* p, int off) {
  u16x16 t = *(const u16x16*)(p + off);
  return __builtin_bit_cast(v16bf, t);
}

// Max-reduction across each 16-lane half-wave, pure VALU (DPP16).
// XOR-mask set {1,2,7,15} generates the full 16-lane group.
__device__ __forceinline__ float rmax16(float x) {
  DPP_MAXF(x, 0xB1);   // quad_perm [1,0,3,2]  : xor 1
  DPP_MAXF(x, 0x4E);   // quad_perm [2,3,0,1]  : xor 2
  DPP_MAXF(x, 0x141);  // row_half_mirror      : xor 7
  DPP_MAXF(x, 0x140);  // row_mirror           : xor 15
  return x;
}

__global__ __launch_bounds__(256, 1)
void flash_attn_causal_f32(const float* __restrict__ q, const float* __restrict__ k,
                           const float* __restrict__ v, float* __restrict__ out) {
  __shared__ __align__(32) unsigned short ldsKT[HEAD_D * BLOCK_N];      // [n][d], 16 KB
  __shared__ __align__(32) unsigned short ldsV [BLOCK_N * HEAD_D];      // [d][n], 16 KB
  __shared__ __align__(32) unsigned short ldsP [NWAVES * 16 * BLOCK_N]; // 16x128/wave, 32 KB

  const int tid  = threadIdx.x;
  const int wave = tid >> 5;
  const int lane = tid & 31;
  const int hi16 = lane >> 4;      // 0: M/K low half, 1: high half
  const int lo16 = lane & 15;

  const size_t bhoff = (size_t)blockIdx.y * T_SEQ * HEAD_D;
  const int qbase = blockIdx.x * BLOCK_M;
  const int row0  = qbase + wave * 16;                     // wave's first query row
  const int row0u = __builtin_amdgcn_readfirstlane(row0);  // scalar for branches
  const int nkb   = (qbase + BLOCK_M) / BLOCK_N;           // blocks up to the diagonal

  // ---- Q tile, register-resident in WMMA A-layout, pre-scaled by 1/sqrt(D) ----
  v16bf aq[2];
  {
    const float* qrow = q + bhoff + (size_t)(row0 + lo16) * HEAD_D;
    const int c = hi16 * 8;
    #pragma unroll
    for (int j = 0; j < 2; ++j) {                 // j: D-halves 0..31 / 32..63
      f32x8 qa = *(const f32x8*)(qrow + 32 * j + c) * 0.125f;
      f32x8 qb = *(const f32x8*)(qrow + 32 * j + c + 16) * 0.125f;
      v8bf ha = __builtin_convertvector(qa, v8bf);
      v8bf hb = __builtin_convertvector(qb, v8bf);
      aq[j] = __builtin_shufflevector(ha, hb, 0,1,2,3,4,5,6,7,8,9,10,11,12,13,14,15);
    }
  }

  // Constant all-ones B operand: rowsum(P) via the matrix engine.
  v16bf bones;
  {
    u16x16 t;
    #pragma unroll
    for (int u = 0; u < 16; ++u) t[u] = 0x3F80;   // bf16(1.0)
    bones = __builtin_bit_cast(v16bf, t);
  }

  const v8f z8 = {0.f, 0.f, 0.f, 0.f, 0.f, 0.f, 0.f, 0.f};
  v8f acc[4] = {z8, z8, z8, z8};                  // 16 x 64 output accumulator
  v8f accL   = z8;                                // softmax denominator (same C-layout)
  float mrow[8];
  #pragma unroll
  for (int i = 0; i < 8; ++i) mrow[i] = NEG_BIG;

  const int frow = tid >> 2;                      // 0..63 : base key row to stage
  const int fseg = (tid & 3) * 16;                // 16-float segment of that row

  for (int kb = 0; kb < nkb; ++kb) {
    const int kstart = kb * BLOCK_N;

    // ---- cooperative K/V staging: vector loads, vector f32->bf16 converts ----
    #pragma unroll
    for (int half = 0; half < 2; ++half) {        // each thread stages 2 rows' segment
      const int r = frow + 64 * half;
      const float* krow = k + bhoff + (size_t)(kstart + r) * HEAD_D + fseg;
      const float* vrow = v + bhoff + (size_t)(kstart + r) * HEAD_D + fseg;
      f32x16 kf = *(const f32x16*)krow;           // 4x global_load_b128
      f32x16 vf = *(const f32x16*)vrow;
      u16x16 kh = __builtin_bit_cast(u16x16, __builtin_convertvector(kf, v16bf));
      u16x16 vh = __builtin_bit_cast(u16x16, __builtin_convertvector(vf, v16bf));
      *(u16x16*)(ldsKT + r * HEAD_D + fseg) = kh; // one 32B store
      #pragma unroll
      for (int u = 0; u < 16; ++u)
        ldsV[(fseg + u) * BLOCK_N + r] = vh[u];   // transposed scatter
      if (kb + 1 < nkb) {                         // global_prefetch_b8 of next block
        __builtin_prefetch(krow + BLOCK_N * HEAD_D, 0, 1);
        __builtin_prefetch(vrow + BLOCK_N * HEAD_D, 0, 1);
      }
    }
    __syncthreads();

    if (kstart <= row0u + 15) {                   // scalar branch: EXEC all-ones
      // ---- S = Q * K^T for the whole 128-key block (eight 16-key N-subtiles) ----
      v8f s[8];
      const int c = hi16 * 8;
      #pragma unroll
      for (int n = 0; n < 8; ++n) {
        v16bf blo = lds_a_gather(ldsKT + (n * 16 + lo16) * HEAD_D, c, c + 16);
        v16bf bhi = lds_a_gather(ldsKT + (n * 16 + lo16) * HEAD_D, 32 + c, 48 + c);
        s[n] = WMMA_BF16(aq[0], blo, z8);
        s[n] = WMMA_BF16(aq[1], bhi, s[n]);
      }

      // ---- causal mask (diagonal block only; scalar condition) ----
      if (kstart + BLOCK_N - 1 > row0u) {
        const int colb = kstart + lo16;
        const int rb   = row0 + hi16 * 8;
        #pragma unroll
        for (int n = 0; n < 8; ++n)
          #pragma unroll
          for (int i = 0; i < 8; ++i)
            if (colb + 16 * n > rb + i) s[n][i] = NEG_BIG;
      }

      // ---- one online-softmax update for all 128 keys ----
      float sc[8];
      #pragma unroll
      for (int i = 0; i < 8; ++i) {
        float mx = fmaxf(fmaxf(fmaxf(s[0][i], s[1][i]), fmaxf(s[2][i], s[3][i])),
                         fmaxf(fmaxf(s[4][i], s[5][i]), fmaxf(s[6][i], s[7][i])));
        mx = rmax16(mx);
        float mn = fmaxf(mrow[i], mx);
        sc[i]   = __expf(mrow[i] - mn);
        mrow[i] = mn;
      }
      #pragma unroll
      for (int n = 0; n < 8; ++n)
        #pragma unroll
        for (int i = 0; i < 8; ++i)
          s[n][i] = __expf(s[n][i] - mrow[i]);
      #pragma unroll
      for (int dg = 0; dg < 4; ++dg)
        #pragma unroll
        for (int i = 0; i < 8; ++i) acc[dg][i] *= sc[i];
      #pragma unroll
      for (int i = 0; i < 8; ++i) accL[i] *= sc[i];

      // ---- P (C-layout) -> per-wave LDS -> four A-operands (K=128) ----
      unsigned short* pP = ldsP + wave * (16 * BLOCK_N);
      #pragma unroll
      for (int n = 0; n < 8; ++n)
        #pragma unroll
        for (int i = 0; i < 8; ++i)
          pP[(i + 8 * hi16) * BLOCK_N + 16 * n + lo16] = f2bf(s[n][i]);
      asm volatile("s_wait_dscnt 0" ::: "memory");    // cross-lane LDS ordering

      v16bf pa[4];
      #pragma unroll
      for (int t = 0; t < 4; ++t)                     // keys 32t .. 32t+31
        pa[t] = lds_a_gather(pP + lo16 * BLOCK_N, 32 * t + c, 32 * t + c + 16);

      // ---- O += P * V (K=128: four chained WMMAs per d-group) + denominator ----
      #pragma unroll
      for (int dg = 0; dg < 4; ++dg) {
        #pragma unroll
        for (int t = 0; t < 4; ++t) {
          v16bf bv = lds_a_gather(ldsV + (dg * 16 + lo16) * BLOCK_N, 32 * t + c, 32 * t + c + 16);
          acc[dg] = WMMA_BF16(pa[t], bv, acc[dg]);
        }
      }
      #pragma unroll
      for (int t = 0; t < 4; ++t)
        accL = WMMA_BF16(pa[t], bones, accL);         // l += rowsum(P)
    }
    __syncthreads();
  }

  // ---- epilogue: normalize and store fp32 ----
  float rinv[8];
  #pragma unroll
  for (int i = 0; i < 8; ++i) rinv[i] = 1.0f / accL[i];
  #pragma unroll
  for (int dg = 0; dg < 4; ++dg)
    #pragma unroll
    for (int i = 0; i < 8; ++i)
      out[bhoff + (size_t)(row0 + i + 8 * hi16) * HEAD_D + dg * 16 + lo16] =
          acc[dg][i] * rinv[i];
}

extern "C" void kernel_launch(void* const* d_in, const int* in_sizes, int n_in,
                              void* d_out, int out_size, void* d_ws, size_t ws_size,
                              hipStream_t stream) {
  (void)in_sizes; (void)n_in; (void)d_ws; (void)ws_size; (void)out_size;
  const float* q = (const float*)d_in[0];
  const float* k = (const float*)d_in[1];
  const float* v = (const float*)d_in[2];
  float* out = (float*)d_out;
  dim3 grid(T_SEQ / BLOCK_M, 2 * 12);   // 16 q-blocks x (B*H)=24
  flash_attn_causal_f32<<<grid, dim3(256), 0, stream>>>(q, k, v, out);
}
